// TransEncoderDecoder_34059090657399
// MI455X (gfx1250) — matrix operand order repaired and checked by reading, not compiled, not run on hardware
//
#include <hip/hip_runtime.h>
#include <math.h>

// ---------------- model constants ----------------
#define BB    2
#define TT    1024
#define CC    1024
#define NHD   16
#define HS    64
#define HHALF 32
#define NKVH  4
#define QPK   4
#define II    2816
#define VV    32000
#define NENC  8
#define NDEC  4
#define BLK   128
#define EPSV  1e-5f
#define SCL   0.125f      // 1/sqrt(64)
#define QKVD  1536        // (NH + 2*NKV) * HS

typedef unsigned short u16;
typedef __attribute__((ext_vector_type(16))) __bf16         v16bf;
typedef __attribute__((ext_vector_type(8)))  float          v8f;
typedef __attribute__((ext_vector_type(4)))  unsigned int   u32x4;
typedef __attribute__((ext_vector_type(4)))  float          f32x4;
typedef __attribute__((ext_vector_type(4)))  unsigned short u16x4;

__device__ __forceinline__ u16 f2bf(float f) {
  unsigned int u = __builtin_bit_cast(unsigned int, f);
  unsigned int r = u + 0x7FFFu + ((u >> 16) & 1u);   // round-to-nearest-even
  return (u16)(r >> 16);
}

// Async copy 16B global -> LDS (CDNA5 GLOBAL_LOAD_ASYNC_TO_LDS_B128, ASYNCcnt).
// lds_byte: raw LDS byte address (= low 32 bits of the flat address of a
// __shared__ object, per the ISA LDS-aperture rule).
__device__ __forceinline__ void async_ld16(unsigned lds_byte, const u16* g) {
  asm volatile("global_load_async_to_lds_b128 %0, %1, off"
               :: "v"(lds_byte), "v"(g) : "memory");
}
__device__ __forceinline__ void wait_async0() {
  asm volatile("s_wait_asynccnt 0x0" ::: "memory");
}
__device__ __forceinline__ unsigned lds_addr(const void* p) {
  return (unsigned)(unsigned long long)p;   // flat addr low 32 bits = LDS offset
}

// =====================================================================
// WMMA GEMM:  C[M,N] = A[M,K] @ W[N,K]^T, A/W bf16 (u16), C f32.
// Requires M%128==0, N%64==0, K%64==0 (true for every GEMM here).
// 256 threads = 8 waves. Block tile 128x64, K-step 64, double-buffered LDS
// filled by GLOBAL_LOAD_ASYNC_TO_LDS_B128. Wave (m=wave&3, n=wave>>2)
// computes a 32x32 tile = 4 accumulators => 8 WMMAs per K-step, with each
// A/B fragment (2x ds_load_b128) reused twice.
// =====================================================================
#define LP 72            // LDS row pitch in u16 (+8 pad, 144B rows)

__global__ __launch_bounds__(256)
void gemm_wmma_bf16(const u16* __restrict__ A, const u16* __restrict__ W,
                    float* __restrict__ Cmat, int M, int N, int Kdim) {
  __shared__ __align__(16) u16 As[2][128][LP];
  __shared__ __align__(16) u16 Bs[2][64][LP];

  const int tid  = threadIdx.x;
  const int wave = tid >> 5;
  const int lane = tid & 31;
  const int half = lane >> 4;            // 0 or 1
  const int lr   = lane & 15;
  const int m0   = (wave & 3) * 32;      // wave row offset in 128-tile
  const int n0   = (wave >> 2) * 32;     // wave col offset in 64-tile

  const long row0 = (long)blockIdx.y * 128;
  const long col0 = (long)blockIdx.x * 64;

  // cooperative load mapping
  const int ra = tid >> 1;               // 0..127, A row
  const int ka = (tid & 1) * 32;         // 0 / 32 : 32 u16 = 4x b128
  const int rb = tid >> 2;               // 0..63, W row (N dim)
  const int kb = (tid & 3) * 16;         // 0..48 : 16 u16 = 2x b128

  const u16* Ag = A + (row0 + ra) * (long)Kdim + ka;
  const u16* Bg = W + (col0 + rb) * (long)Kdim + kb;

  const unsigned aoff[2] = { lds_addr(&As[0][ra][ka]), lds_addr(&As[1][ra][ka]) };
  const unsigned boff[2] = { lds_addr(&Bs[0][rb][kb]), lds_addr(&Bs[1][rb][kb]) };

  auto issue = [&](int buf, int kk) {
    const u16* ag = Ag + kk;
    const u16* bg = Bg + kk;
    #pragma unroll
    for (int j = 0; j < 4; ++j) async_ld16(aoff[buf] + j * 16, ag + j * 8);
    async_ld16(boff[buf],      bg);
    async_ld16(boff[buf] + 16, bg + 8);
  };

  v8f acc00 = {}, acc01 = {}, acc10 = {}, acc11 = {};

  issue(0, 0);
  wait_async0();
  __syncthreads();

  int buf = 0;
  for (int kk = 0; kk < Kdim; kk += 64) {
    if (kk + 64 < Kdim) issue(buf ^ 1, kk + 64);

    #pragma unroll
    for (int ks = 0; ks < 2; ++ks) {     // two 32-wide K chunks
      const int kc = ks * 32;
      union { u32x4 u[2]; v16bf v; } fa0, fa1, fb0, fb1;
      // A fragment: lanes 0-15 -> K={0..7,16..23}; lanes 16-31 -> K={8..15,24..31}
      {
        const u16* p = &As[buf][m0 + lr][kc + half * 8];
        fa0.u[0] = *(const u32x4*)p;
        fa0.u[1] = *(const u32x4*)(p + 16);
        const u16* q = &As[buf][m0 + 16 + lr][kc + half * 8];
        fa1.u[0] = *(const u32x4*)q;
        fa1.u[1] = *(const u32x4*)(q + 16);
      }
      // B fragment: lanes 0-15 -> K={0..7,16..23}; lanes 16-31 -> K={8..15,24..31}
      {
        const u16* p = &Bs[buf][n0 + lr][kc + half * 8];
        fb0.u[0] = *(const u32x4*)p;
        fb0.u[1] = *(const u32x4*)(p + 16);
        const u16* q = &Bs[buf][n0 + 16 + lr][kc + half * 8];
        fb1.u[0] = *(const u32x4*)q;
        fb1.u[1] = *(const u32x4*)(q + 16);
      }
      acc00 = __builtin_amdgcn_wmma_f32_16x16x32_bf16(false, fa0.v, false, fb0.v, (short)0, acc00, false, false);
      acc01 = __builtin_amdgcn_wmma_f32_16x16x32_bf16(false, fa0.v, false, fb1.v, (short)0, acc01, false, false);
      acc10 = __builtin_amdgcn_wmma_f32_16x16x32_bf16(false, fa1.v, false, fb0.v, (short)0, acc10, false, false);
      acc11 = __builtin_amdgcn_wmma_f32_16x16x32_bf16(false, fa1.v, false, fb1.v, (short)0, acc11, false, false);
    }

    wait_async0();        // next buffer's async fills complete (this wave)
    __syncthreads();      // all waves done reading current buffer
    buf ^= 1;
  }

  // C/D layout: lanes 0-15: vgpr r -> M=r; lanes 16-31: vgpr r -> M=r+8; N=lr
  const long rbase = row0 + m0 + half * 8;
  const long cbase = col0 + n0 + lr;
  #pragma unroll
  for (int r = 0; r < 8; ++r) {
    Cmat[(rbase + r) * (long)N + cbase]           = acc00[r];
    Cmat[(rbase + r) * (long)N + cbase + 16]      = acc01[r];
    Cmat[(rbase + 16 + r) * (long)N + cbase]      = acc10[r];
    Cmat[(rbase + 16 + r) * (long)N + cbase + 16] = acc11[r];
  }
}

// =====================================================================
// f32 -> bf16 conversion (float4 in, ushort4 out)
// =====================================================================
__global__ void cvt_bf16_k(const float* __restrict__ src, u16* __restrict__ dst, long n4) {
  long i = (long)blockIdx.x * 256 + threadIdx.x;
  if (i >= n4) return;
  f32x4 f = ((const f32x4*)src)[i];
  u16x4 o;
  o.x = f2bf(f.x); o.y = f2bf(f.y); o.z = f2bf(f.z); o.w = f2bf(f.w);
  ((u16x4*)dst)[i] = o;
}

// =====================================================================
// Embedding gather
// =====================================================================
__global__ void embed_k(const float* __restrict__ wte, const int* __restrict__ idx,
                        float* __restrict__ out, long ntok) {
  long i = (long)blockIdx.x * 256 + threadIdx.x;
  long tot = ntok * CC;
  if (i >= tot) return;
  long tok = i / CC; int c = (int)(i % CC);
  out[i] = wte[(long)idx[tok] * CC + c];
}

// =====================================================================
// RMSNorm (one block / row). _bf16 variant feeds GEMMs.
// =====================================================================
__device__ __forceinline__ float rms_scale(const float* xp, int Cdim, float* red) {
  float ss = 0.f;
  for (int c = threadIdx.x; c < Cdim; c += 256) { float v = xp[c]; ss += v * v; }
  #pragma unroll
  for (int off = 16; off >= 1; off >>= 1) ss += __shfl_xor(ss, off, 32);
  if ((threadIdx.x & 31) == 0) red[threadIdx.x >> 5] = ss;
  __syncthreads();
  if (threadIdx.x == 0) {
    float t = 0.f;
    #pragma unroll
    for (int i = 0; i < 8; ++i) t += red[i];
    red[0] = rsqrtf(t / (float)Cdim + EPSV);
  }
  __syncthreads();
  return red[0];
}

__global__ __launch_bounds__(256)
void rmsnorm_f32_k(const float* __restrict__ x, const float* __restrict__ w,
                   float* __restrict__ y, int Cdim) {
  __shared__ float red[8];
  long row = blockIdx.x;
  const float* xp = x + row * (long)Cdim;
  float sc = rms_scale(xp, Cdim, red);
  float* yp = y + row * (long)Cdim;
  for (int c = threadIdx.x; c < Cdim; c += 256) yp[c] = xp[c] * sc * w[c];
}

__global__ __launch_bounds__(256)
void rmsnorm_bf16_k(const float* __restrict__ x, const float* __restrict__ w,
                    u16* __restrict__ y, int Cdim) {
  __shared__ float red[8];
  long row = blockIdx.x;
  const float* xp = x + row * (long)Cdim;
  float sc = rms_scale(xp, Cdim, red);
  u16* yp = y + row * (long)Cdim;
  for (int c = threadIdx.x; c < Cdim; c += 256) yp[c] = f2bf(xp[c] * sc * w[c]);
}

// =====================================================================
// Pack kernels: GEMM outputs -> [B][H][T][HS] head-major f32
// =====================================================================
__global__ void pack_enc_qkv(const float* __restrict__ src, float* __restrict__ Qp,
                             float* __restrict__ Kp, float* __restrict__ Vp,
                             int Bn, int Tn) {
  long i = (long)blockIdx.x * 256 + threadIdx.x;
  long tot = (long)Bn * Tn * QKVD;
  if (i >= tot) return;
  int col = (int)(i % QKVD); long tok = i / QKVD;
  int b = (int)(tok / Tn), t = (int)(tok % Tn);
  int g  = col / ((QPK + 2) * HS);
  int r  = (col / HS) % (QPK + 2);
  int dd = col % HS;
  float v = src[i];
  if (r < QPK)
    Qp[(((long)b * NHD + g * QPK + r) * Tn + t) * HS + dd] = v;
  else if (r == QPK)
    Kp[(((long)b * NKVH + g) * Tn + t) * HS + dd] = v;
  else
    Vp[(((long)b * NKVH + g) * Tn + t) * HS + dd] = v;
}

__global__ void pack_heads(const float* __restrict__ src, float* __restrict__ Qp,
                           int Bn, int Tn) {
  long i = (long)blockIdx.x * 256 + threadIdx.x;
  long tot = (long)Bn * Tn * CC;
  if (i >= tot) return;
  int col = (int)(i % CC); long tok = i / CC;
  int b = (int)(tok / Tn), t = (int)(tok % Tn);
  int head = col / HS, dd = col % HS;
  Qp[(((long)b * NHD + head) * Tn + t) * HS + dd] = src[i];
}

__global__ void pack_kv(const float* __restrict__ src, float* __restrict__ Kp,
                        float* __restrict__ Vp, int Bn, int Tn) {
  long i = (long)blockIdx.x * 256 + threadIdx.x;
  long tot = (long)Bn * Tn * (2 * NKVH * HS);
  if (i >= tot) return;
  int col = (int)(i % (2 * NKVH * HS)); long tok = i / (2 * NKVH * HS);
  int b = (int)(tok / Tn), t = (int)(tok % Tn);
  int g = col / (2 * HS);
  int r = (col / HS) & 1;
  int dd = col % HS;
  float* dst = r ? Vp : Kp;
  dst[(((long)b * NKVH + g) * Tn + t) * HS + dd] = src[i];
}

// =====================================================================
// RoPE in-place on [B][H][T][HS]
// =====================================================================
__global__ void rope_k(float* __restrict__ X, int Bn, int H, int Tn, int posOff) {
  long i = (long)blockIdx.x * 256 + threadIdx.x;
  long tot = (long)Bn * H * Tn * HHALF;
  if (i >= tot) return;
  int hi = (int)(i % HHALF);
  long rest = i / HHALF;
  int t = (int)(rest % Tn);
  float* p = X + rest * HS;
  float x1 = p[hi], x2 = p[hi + HHALF];
  float inv = __expf(-((2.0f * hi) / (float)HS) * logf(10000.0f));
  float fr = (float)(t + posOff) * inv;
  float c = cosf(fr), s = sinf(fr);
  p[hi]         = x1 * c - x2 * s;
  p[hi + HHALF] = x2 * c + x1 * s;
}

// =====================================================================
// Attention: one wave per (b, head, q), online softmax, block-sparse mask.
// Output bf16 (feeds the proj GEMM).
// mode 0 (enc): mask = block(k) >= block(q), Tk == Tq
// mode 1 (dec): k<Tq -> block(q) > block(k); else block(q) == block(k-Tq)
// =====================================================================
__global__ __launch_bounds__(128)
void attn_k(const float* __restrict__ Q, const float* __restrict__ Kc,
            const float* __restrict__ Vc, u16* __restrict__ O,
            int Bn, int Tq, int Tk, int mode) {
  int gw = blockIdx.x * 4 + (threadIdx.x >> 5);
  int lane = threadIdx.x & 31;
  int q = gw % Tq;
  int h = (gw / Tq) % NHD;
  int b = gw / (Tq * NHD);
  if (b >= Bn) return;
  int g = h / QPK;
  const float* qp = Q + (((long)(b * NHD + h) * Tq + q) * HS);
  float q0 = qp[lane * 2], q1 = qp[lane * 2 + 1];
  const float* kbase = Kc + ((long)(b * NKVH + g) * Tk) * HS;
  const float* vbase = Vc + ((long)(b * NKVH + g) * Tk) * HS;
  int qb = q / BLK;
  float m = -1e30f, s = 0.f, a0 = 0.f, a1 = 0.f;
  for (int k = 0; k < Tk; ++k) {
    bool ok;
    if (mode == 0)       ok = (k / BLK) >= qb;
    else if (k < Tq)     ok = qb > (k / BLK);
    else                 ok = qb == ((k - Tq) / BLK);
    if (!ok) continue;
    const float* kp = kbase + (long)k * HS;
    float d = q0 * kp[lane * 2] + q1 * kp[lane * 2 + 1];
    #pragma unroll
    for (int off = 16; off >= 1; off >>= 1) d += __shfl_xor(d, off, 32);
    float sc = d * SCL;
    float nm = fmaxf(m, sc);
    float f  = __expf(m - nm);
    float p  = __expf(sc - nm);
    const float* vp = vbase + (long)k * HS;
    a0 = a0 * f + p * vp[lane * 2];
    a1 = a1 * f + p * vp[lane * 2 + 1];
    s  = s * f + p;
    m  = nm;
  }
  float inv = 1.0f / s;
  u16* op = O + ((long)b * Tq + q) * (long)(NHD * HS) + h * HS;
  op[lane * 2]     = f2bf(a0 * inv);
  op[lane * 2 + 1] = f2bf(a1 * inv);
}

// =====================================================================
// Elementwise helpers
// =====================================================================
__global__ void add_inplace_k(float* __restrict__ y, const float* __restrict__ x, long n) {
  long i = (long)blockIdx.x * 256 + threadIdx.x;
  if (i < n) y[i] += x[i];
}
__global__ void silu_mul_bf16_k(const float* __restrict__ a, const float* __restrict__ b,
                                u16* __restrict__ o, long n) {
  long i = (long)blockIdx.x * 256 + threadIdx.x;
  if (i < n) { float v = a[i]; o[i] = f2bf(v / (1.0f + __expf(-v)) * b[i]); }
}
__global__ void concat_rows_k(const float* __restrict__ a, const float* __restrict__ b,
                              float* __restrict__ out, int Bn, int Tn) {
  long i = (long)blockIdx.x * 256 + threadIdx.x;
  long tot = (long)Bn * 2 * Tn * CC;
  if (i >= tot) return;
  int c = (int)(i % CC);
  long row = i / CC;
  int t2 = (int)(row % (2 * Tn));
  int b_ = (int)(row / (2 * Tn));
  out[i] = (t2 < Tn) ? a[((long)b_ * Tn + t2) * CC + c]
                     : b[((long)b_ * Tn + (t2 - Tn)) * CC + c];
}
__global__ void extract_dec_bf16_k(const u16* __restrict__ h, u16* __restrict__ hd,
                                   int Bn, int Tn) {
  long i = (long)blockIdx.x * 256 + threadIdx.x;
  long tot = (long)Bn * Tn * CC;
  if (i >= tot) return;
  int c = (int)(i % CC);
  long row = i / CC;
  int t = (int)(row % Tn);
  int b = (int)(row / Tn);
  hd[i] = h[((long)b * 2 * Tn + Tn + t) * CC + c];
}

// =====================================================================
// Orchestration
// =====================================================================
static inline dim3 ew_grid(long n) { return dim3((unsigned)((n + 255) / 256)); }

extern "C" void kernel_launch(void* const* d_in, const int* in_sizes, int n_in,
                              void* d_out, int out_size, void* d_ws, size_t ws_size,
                              hipStream_t stream) {
  const float* wte        = (const float*)d_in[0];
  const float* lm_head_w  = (const float*)d_in[1];
  const float* enc_attn_w = (const float*)d_in[2];
  const float* enc_proj_w = (const float*)d_in[3];
  const float* enc_ln1    = (const float*)d_in[4];
  const float* enc_ln2    = (const float*)d_in[5];
  const float* enc_w1     = (const float*)d_in[6];
  const float* enc_w2     = (const float*)d_in[7];
  const float* enc_w3     = (const float*)d_in[8];
  const float* enc_normf  = (const float*)d_in[9];
  const float* dec_q_w    = (const float*)d_in[10];
  const float* dec_kv_w   = (const float*)d_in[11];
  const float* dec_proj_w = (const float*)d_in[12];
  const float* dec_ln1    = (const float*)d_in[13];
  const float* dec_ln2    = (const float*)d_in[14];
  const float* dec_w1     = (const float*)d_in[15];
  const float* dec_w2     = (const float*)d_in[16];
  const float* dec_w3     = (const float*)d_in[17];
  const float* dec_normf  = (const float*)d_in[18];
  const int*   clean_idx  = (const int*)d_in[19];
  const int*   noisy_idx  = (const int*)d_in[20];

  const long ntok  = (long)BB * TT;   // 2048
  const long ntok2 = 2 * ntok;        // 4096

  char* base = (char*)d_ws;
  size_t off = 0;
  auto alloc = [&](size_t bytes) -> void* {
    void* p = base + off;
    off = (off + bytes + 255) & ~(size_t)255;
    return p;
  };
  float* x    = (float*)alloc(ntok  * CC * 4);
  float* qkv  = (float*)alloc(ntok  * QKVD * 4);        // also dec-q out
  float* Qp   = (float*)alloc(ntok  * CC * 4);          // [B][NH][T][HS]
  float* Kp   = (float*)alloc(ntok2 * NKVH * HS * 4);   // sized for dec (Tk=2T)
  float* Vp   = (float*)alloc(ntok2 * NKVH * HS * 4);
  float* m1   = (float*)alloc(ntok  * II * 4);
  float* m2   = (float*)alloc(ntok  * II * 4);
  float* encx = (float*)alloc(ntok  * CC * 4);
  float* dbuf = (float*)alloc(ntok  * CC * 4);
  float* full = (float*)alloc(ntok2 * CC * 4);
  float* kvb  = (float*)alloc(ntok2 * 2 * NKVH * HS * 4);
  u16*  hb    = (u16*)alloc(ntok2 * CC * 2);            // bf16 norm buf
  u16*  attb  = (u16*)alloc(ntok  * CC * 2);            // bf16 attention out
  u16*  mb    = (u16*)alloc(ntok  * II * 2);            // bf16 silu-mul out
  u16*  hdecb = (u16*)alloc(ntok  * CC * 2);
  u16*  wscr  = (u16*)alloc((long)VV * CC * 2);         // bf16 weight scratch

  auto CVT = [&](const float* w, long n) {  // weights f32 -> bf16 scratch
    cvt_bf16_k<<<ew_grid(n / 4), 256, 0, stream>>>(w, wscr, n / 4);
  };
  auto GEMM = [&](const u16* A, const u16* W, float* Cm, long M, long N, long K) {
    gemm_wmma_bf16<<<dim3((unsigned)(N / 64), (unsigned)(M / 128)), dim3(256), 0, stream>>>(
        A, W, Cm, (int)M, (int)N, (int)K);
  };

  // ---------------- Encoder ----------------
  embed_k<<<ew_grid(ntok * CC), 256, 0, stream>>>(wte, clean_idx, x, ntok);

  for (int l = 0; l < NENC; ++l) {
    const float* attw  = enc_attn_w + (long)l * QKVD * CC;
    const float* projw = enc_proj_w + (long)l * CC * CC;
    const float* w1    = enc_w1 + (long)l * II * CC;
    const float* w2    = enc_w2 + (long)l * II * CC;
    const float* w3    = enc_w3 + (long)l * CC * II;

    rmsnorm_bf16_k<<<(unsigned)ntok, 256, 0, stream>>>(x, enc_ln1 + (long)l * CC, hb, CC);
    CVT(attw, (long)QKVD * CC);
    GEMM(hb, wscr, qkv, ntok, QKVD, CC);
    pack_enc_qkv<<<ew_grid(ntok * QKVD), 256, 0, stream>>>(qkv, Qp, Kp, Vp, BB, TT);
    rope_k<<<ew_grid((long)BB * NHD * TT * HHALF), 256, 0, stream>>>(Qp, BB, NHD, TT, 0);
    rope_k<<<ew_grid((long)BB * NKVH * TT * HHALF), 256, 0, stream>>>(Kp, BB, NKVH, TT, 0);
    attn_k<<<(unsigned)(BB * NHD * TT / 4), 128, 0, stream>>>(Qp, Kp, Vp, attb, BB, TT, TT, 0);
    CVT(projw, (long)CC * CC);
    GEMM(attb, wscr, m1, ntok, CC, CC);
    add_inplace_k<<<ew_grid(ntok * CC), 256, 0, stream>>>(x, m1, ntok * CC);

    rmsnorm_bf16_k<<<(unsigned)ntok, 256, 0, stream>>>(x, enc_ln2 + (long)l * CC, hb, CC);
    CVT(w1, (long)II * CC);
    GEMM(hb, wscr, m1, ntok, II, CC);
    CVT(w2, (long)II * CC);
    GEMM(hb, wscr, m2, ntok, II, CC);
    silu_mul_bf16_k<<<ew_grid(ntok * II), 256, 0, stream>>>(m1, m2, mb, ntok * II);
    CVT(w3, (long)CC * II);
    GEMM(mb, wscr, m1, ntok, CC, II);
    add_inplace_k<<<ew_grid(ntok * CC), 256, 0, stream>>>(x, m1, ntok * CC);
  }
  rmsnorm_f32_k<<<(unsigned)ntok, 256, 0, stream>>>(x, enc_normf, encx, CC);

  // ---------------- Decoder ----------------
  embed_k<<<ew_grid(ntok * CC), 256, 0, stream>>>(wte, noisy_idx, dbuf, ntok);

  for (int l = 0; l < NDEC; ++l) {
    const float* qw    = dec_q_w    + (long)l * CC * CC;
    const float* kvw   = dec_kv_w   + (long)l * (2 * NKVH * HS) * CC;
    const float* projw = dec_proj_w + (long)l * CC * CC;
    const float* w1    = dec_w1 + (long)l * II * CC;
    const float* w2    = dec_w2 + (long)l * II * CC;
    const float* w3    = dec_w3 + (long)l * CC * II;

    concat_rows_k<<<ew_grid(ntok2 * CC), 256, 0, stream>>>(encx, dbuf, full, BB, TT);
    rmsnorm_bf16_k<<<(unsigned)ntok2, 256, 0, stream>>>(full, dec_ln1 + (long)l * CC, hb, CC);
    extract_dec_bf16_k<<<ew_grid(ntok * CC), 256, 0, stream>>>(hb, hdecb, BB, TT);

    CVT(qw, (long)CC * CC);
    GEMM(hdecb, wscr, qkv, ntok, CC, CC);                 // q projection
    CVT(kvw, (long)(2 * NKVH * HS) * CC);
    GEMM(hb, wscr, kvb, ntok2, 2 * NKVH * HS, CC);        // kv over full seq
    pack_heads<<<ew_grid(ntok * CC), 256, 0, stream>>>(qkv, Qp, BB, TT);
    pack_kv<<<ew_grid(ntok2 * 2 * NKVH * HS), 256, 0, stream>>>(kvb, Kp, Vp, BB, 2 * TT);
    rope_k<<<ew_grid((long)BB * NHD * TT * HHALF), 256, 0, stream>>>(Qp, BB, NHD, TT, TT);
    rope_k<<<ew_grid((long)BB * NKVH * 2 * TT * HHALF), 256, 0, stream>>>(Kp, BB, NKVH, 2 * TT, 0);
    attn_k<<<(unsigned)(BB * NHD * TT / 4), 128, 0, stream>>>(Qp, Kp, Vp, attb, BB, TT, 2 * TT, 1);
    CVT(projw, (long)CC * CC);
    GEMM(attb, wscr, m1, ntok, CC, CC);
    add_inplace_k<<<ew_grid(ntok * CC), 256, 0, stream>>>(dbuf, m1, ntok * CC);

    rmsnorm_bf16_k<<<(unsigned)ntok, 256, 0, stream>>>(dbuf, dec_ln2 + (long)l * CC, hdecb, CC);
    CVT(w1, (long)II * CC);
    GEMM(hdecb, wscr, m1, ntok, II, CC);
    CVT(w2, (long)II * CC);
    GEMM(hdecb, wscr, m2, ntok, II, CC);
    silu_mul_bf16_k<<<ew_grid(ntok * II), 256, 0, stream>>>(m1, m2, mb, ntok * II);
    CVT(w3, (long)CC * II);
    GEMM(mb, wscr, m1, ntok, CC, II);
    add_inplace_k<<<ew_grid(ntok * CC), 256, 0, stream>>>(dbuf, m1, ntok * CC);
  }

  rmsnorm_bf16_k<<<(unsigned)ntok, 256, 0, stream>>>(dbuf, dec_normf, hdecb, CC);
  CVT(lm_head_w, (long)VV * CC);
  GEMM(hdecb, wscr, (float*)d_out, ntok, VV, CC);         // logits
}
